// PhysicsAwareSpatialTransformer_41205916238345
// MI455X (gfx1250) — matrix operand order repaired and checked by reading, not compiled, not run on hardware
//
#include <hip/hip_runtime.h>
#include <math.h>

typedef _Float16 f16;
typedef __attribute__((ext_vector_type(16))) _Float16 v16h;
typedef __attribute__((ext_vector_type(8)))  _Float16 v8h;
typedef __attribute__((ext_vector_type(8)))  float    v8f;

#define LN_EPSF 1e-6f

// ---------------------------------------------------------------------------
// WMMA helpers (CDNA5 gfx1250, wave32, 16x16x32 f16 -> f32)
// ---------------------------------------------------------------------------
__device__ __forceinline__ v8f wmma32(v16h a, v16h b, v8f c) {
  return __builtin_amdgcn_wmma_f32_16x16x32_f16(false, a, false, b, (short)0, c,
                                                false, false);
}

__device__ __forceinline__ v16h negv(v16h x) {
  v16h r;
#pragma unroll
  for (int i = 0; i < 16; ++i) r[i] = -x[i];
  return r;
}

// A operand: 16x32 f16 tile, row-major in memory (ld = elements per row).
// ISA layout: lane = M + 16*((k&15)>>3); halves 0..7 <-> k = hs*8+0..7,
//             halves 8..15 <-> k = 16 + hs*8 + 0..7
__device__ __forceinline__ v16h load_frag_a(const f16* base, int ld, int lane) {
  const int row = lane & 15;
  const int hs  = lane >> 4;
  const f16* p = base + (size_t)row * ld + hs * 8;
  v8h c0 = *(const v8h*)(p);
  v8h c1 = *(const v8h*)(p + 16);
  v16h r;
#pragma unroll
  for (int i = 0; i < 8; ++i) { r[i] = c0[i]; r[i + 8] = c1[i]; }
  return r;
}

// B operand: 32x16 f16 tile; we feed B^T rows (output column n), row-major
// with ld elements per row, reading K contiguous.
// ISA layout: lanes 0-15 <-> N; halves 0..7 <-> k = ks*8+0..7,
// halves 8..15 <-> k = 16 + ks*8 + 0..7 (same as A).
__device__ __forceinline__ v16h load_frag_b(const f16* base, int ld, int lane) {
  const int col = lane & 15;
  const int ks  = lane >> 4;
  const f16* p = base + (size_t)col * ld + ks * 8;
  v8h c0 = *(const v8h*)(p);
  v8h c1 = *(const v8h*)(p + 16);
  v16h r;
#pragma unroll
  for (int i = 0; i < 8; ++i) { r[i] = c0[i]; r[i + 8] = c1[i]; }
  return r;
}

__device__ __forceinline__ float gelu_exact(float x) {
  return 0.5f * x * (1.0f + erff(x * 0.70710678118654752f));
}

// ---------------------------------------------------------------------------
// fp32 -> f16 converts (plain, and with pre-negated copy for imag weights)
// ---------------------------------------------------------------------------
__global__ __launch_bounds__(256) void f32_to_f16_kernel(
    const float* __restrict__ in, f16* __restrict__ out, long n) {
  long i = (long)blockIdx.x * 256 + threadIdx.x;
  if (i < n) out[i] = (f16)in[i];
}

__global__ __launch_bounds__(256) void f32_to_f16_neg_kernel(
    const float* __restrict__ in, f16* __restrict__ out,
    f16* __restrict__ outn, long n) {
  long i = (long)blockIdx.x * 256 + threadIdx.x;
  if (i < n) {
    float v = in[i];
    out[i]  = (f16)v;
    outn[i] = (f16)(-v);
  }
}

// ---------------------------------------------------------------------------
// Complex magnitude LayerNorm. One block (256 thr) per row of E=768 features.
// inter!=0: read interleaved x[row, 2E]; else read xr_in/xi_in [row, E].
// Writes f16 normalized (oR16/oI16); optionally raw fp32 copies (rawR/rawI).
// ---------------------------------------------------------------------------
__global__ __launch_bounds__(256) void cln_kernel(
    const float* __restrict__ xin,
    const float* __restrict__ xr_in, const float* __restrict__ xi_in,
    const float* __restrict__ gamma, const float* __restrict__ beta,
    f16* __restrict__ oR16, f16* __restrict__ oI16,
    float* __restrict__ rawR, float* __restrict__ rawI,
    int E, int inter) {
  const int row = blockIdx.x;
  const int tid = threadIdx.x;
  float r[3], im[3], mg[3];     // E/256 == 3
  float s1 = 0.f, s2 = 0.f;
  int cnt = 0;
  for (int j = tid; j < E; j += 256, ++cnt) {
    float a, b;
    if (inter) {
      a = xin[(size_t)row * 2 * E + 2 * j];
      b = xin[(size_t)row * 2 * E + 2 * j + 1];
    } else {
      a = xr_in[(size_t)row * E + j];
      b = xi_in[(size_t)row * E + j];
    }
    r[cnt] = a; im[cnt] = b;
    float m = sqrtf(a * a + b * b + LN_EPSF);
    mg[cnt] = m; s1 += m; s2 += m * m;
  }
  for (int sm = 1; sm < 32; sm <<= 1) {
    s1 += __shfl_xor(s1, sm, 32);
    s2 += __shfl_xor(s2, sm, 32);
  }
  __shared__ float p1[8], p2[8];
  if ((tid & 31) == 0) { p1[tid >> 5] = s1; p2[tid >> 5] = s2; }
  __syncthreads();
  float t1 = 0.f, t2 = 0.f;
#pragma unroll
  for (int w = 0; w < 8; ++w) { t1 += p1[w]; t2 += p2[w]; }
  const float mean = t1 / (float)E;
  const float var  = t2 / (float)E - mean * mean;
  const float rstd = rsqrtf(var + LN_EPSF);
  cnt = 0;
  for (int j = tid; j < E; j += 256, ++cnt) {
    float scale = ((mg[cnt] - mean) * rstd) / (mg[cnt] + LN_EPSF);
    float orv = gamma[2 * j]     * (r[cnt]  * scale) + beta[2 * j];
    float oiv = gamma[2 * j + 1] * (im[cnt] * scale) + beta[2 * j + 1];
    const size_t o = (size_t)row * E + j;
    oR16[o] = (f16)orv;
    oI16[o] = (f16)oiv;
    if (rawR) { rawR[o] = r[cnt]; rawI[o] = im[cnt]; }
  }
}

// ---------------------------------------------------------------------------
// Complex GEMM: out = act @ W^T (+bias) [+GELU] [+residual]
// act f16 [M,K], W f16 [Nout,K], wIn = pre-negated imag weights (-Wi).
// One wave per 32x32 macro tile (2x2 WMMA tiles, 16 WMMA per k-step).
// Two accumulators per tile; no in-loop VALU (avoids WMMA->VALU WAR NOPs):
//   accR += Ar.Br ; accR += Ai.(-Bi) ; accI += Ar.Bi ; accI += Ai.Br
// ---------------------------------------------------------------------------
__global__ __launch_bounds__(128) void cgemm_kernel(
    const f16* __restrict__ aR, const f16* __restrict__ aI,
    const f16* __restrict__ wR, const f16* __restrict__ wI,
    const f16* __restrict__ wIn,
    const float* __restrict__ bR, const float* __restrict__ bI,
    const float* __restrict__ resR, const float* __restrict__ resI,
    float* __restrict__ oR32, float* __restrict__ oI32,
    f16* __restrict__ oR16, f16* __restrict__ oI16,
    float* __restrict__ oInter,
    int M, int K, int Nout, int doGelu) {
  const int lane = threadIdx.x & 31;
  const int wave = threadIdx.x >> 5;
  const int tilesN = Nout >> 5;                 // 32-wide macro cols
  const int tile = blockIdx.x * 4 + wave;
  if (tile >= (M >> 5) * tilesN) return;
  const int tm = tile / tilesN;
  const int tn = tile % tilesN;

  v8f accR[2][2] = {}, accI[2][2] = {};
  const f16* aRp  = aR  + (size_t)tm * 32 * K;
  const f16* aIp  = aI  + (size_t)tm * 32 * K;
  const f16* wRp  = wR  + (size_t)tn * 32 * K;
  const f16* wIp  = wI  + (size_t)tn * 32 * K;
  const f16* wInp = wIn + (size_t)tn * 32 * K;
  const size_t rowK = (size_t)16 * K;

  for (int k = 0; k < K; k += 32) {
    v16h Ar0 = load_frag_a(aRp + k,        K, lane);
    v16h Ar1 = load_frag_a(aRp + rowK + k, K, lane);
    v16h Ai0 = load_frag_a(aIp + k,        K, lane);
    v16h Ai1 = load_frag_a(aIp + rowK + k, K, lane);
    v16h Br0 = load_frag_b(wRp + k,        K, lane);
    v16h Br1 = load_frag_b(wRp + rowK + k, K, lane);
    v16h Bi0 = load_frag_b(wIp + k,        K, lane);
    v16h Bi1 = load_frag_b(wIp + rowK + k, K, lane);
    v16h nBi0 = load_frag_b(wInp + k,        K, lane);
    v16h nBi1 = load_frag_b(wInp + rowK + k, K, lane);

    accR[0][0] = wmma32(Ar0, Br0,  accR[0][0]);
    accR[0][0] = wmma32(Ai0, nBi0, accR[0][0]);
    accI[0][0] = wmma32(Ar0, Bi0,  accI[0][0]);
    accI[0][0] = wmma32(Ai0, Br0,  accI[0][0]);

    accR[0][1] = wmma32(Ar0, Br1,  accR[0][1]);
    accR[0][1] = wmma32(Ai0, nBi1, accR[0][1]);
    accI[0][1] = wmma32(Ar0, Bi1,  accI[0][1]);
    accI[0][1] = wmma32(Ai0, Br1,  accI[0][1]);

    accR[1][0] = wmma32(Ar1, Br0,  accR[1][0]);
    accR[1][0] = wmma32(Ai1, nBi0, accR[1][0]);
    accI[1][0] = wmma32(Ar1, Bi0,  accI[1][0]);
    accI[1][0] = wmma32(Ai1, Br0,  accI[1][0]);

    accR[1][1] = wmma32(Ar1, Br1,  accR[1][1]);
    accR[1][1] = wmma32(Ai1, nBi1, accR[1][1]);
    accI[1][1] = wmma32(Ar1, Bi1,  accI[1][1]);
    accI[1][1] = wmma32(Ai1, Br1,  accI[1][1]);
  }

  const int hs = lane >> 4, nn = lane & 15;
#pragma unroll
  for (int ni = 0; ni < 2; ++ni) {
    const int n = tn * 32 + ni * 16 + nn;
    const float brv = bR ? bR[n] : 0.0f;
    const float biv = bI ? bI[n] : 0.0f;
#pragma unroll
    for (int mi = 0; mi < 2; ++mi) {
#pragma unroll
      for (int e = 0; e < 8; ++e) {
        const int m = tm * 32 + mi * 16 + e + hs * 8;
        float vr = accR[mi][ni][e] + brv;
        float vi = accI[mi][ni][e] + biv;
        if (doGelu) { vr = gelu_exact(vr); vi = gelu_exact(vi); }
        const size_t o = (size_t)m * Nout + n;
        if (resR) { vr += resR[o]; vi += resI[o]; }
        if (oR32) { oR32[o] = vr; oI32[o] = vi; }
        if (oR16) { oR16[o] = (f16)vr; oI16[o] = (f16)vi; }
        if (oInter) {
          oInter[(size_t)m * (2 * Nout) + 2 * n]     = vr;
          oInter[(size_t)m * (2 * Nout) + 2 * n + 1] = vi;
        }
      }
    }
  }
}

// ---------------------------------------------------------------------------
// Split qkv fp32 [B,N,3,nH,hd] into f16 q,k [bh,N,hd] and vT [bh,hd,N]
// ---------------------------------------------------------------------------
__global__ __launch_bounds__(256) void split_qkv_kernel(
    const float* __restrict__ qkvR, const float* __restrict__ qkvI,
    f16* __restrict__ q_r, f16* __restrict__ q_i,
    f16* __restrict__ k_r, f16* __restrict__ k_i,
    f16* __restrict__ vT_r, f16* __restrict__ vT_i) {
  long idx = (long)blockIdx.x * 256 + threadIdx.x;   // B*N*nH*hd
  const int d = (int)(idx & 63);
  long t = idx >> 6;
  const int h = (int)(t % 12); t /= 12;
  const int n = (int)(t % 2048);
  const int b = (int)(t / 2048);
  const size_t src = ((size_t)(b * 2048 + n)) * 2304 + h * 64 + d;
  const size_t dq  = ((size_t)((b * 12 + h) * 2048 + n)) * 64 + d;
  const size_t dv  = ((size_t)((b * 12 + h) * 64 + d)) * 2048 + n;
  q_r[dq]  = (f16)qkvR[src];          q_i[dq]  = (f16)qkvI[src];
  k_r[dq]  = (f16)qkvR[src + 768];    k_i[dq]  = (f16)qkvI[src + 768];
  vT_r[dv] = (f16)qkvR[src + 1536];   vT_i[dv] = (f16)qkvI[src + 1536];
}

// ---------------------------------------------------------------------------
// Flash-style complex attention. One wave per (bh, 16-query tile).
// Sr = qr.krT + qi.kiT ; Si = qr.kiT + (-qi).krT  (negated-qi fragments are
// loop-invariant). mag = sqrt(Sr^2+Si^2+1e-8)/8; online softmax; P (f16 via
// per-wave LDS transpose) x V accumulation.
// ---------------------------------------------------------------------------
__global__ __launch_bounds__(128) void cattn_kernel(
    const f16* __restrict__ qr, const f16* __restrict__ qi,
    const f16* __restrict__ kr, const f16* __restrict__ ki,
    const f16* __restrict__ vTr, const f16* __restrict__ vTi,
    f16* __restrict__ outR, f16* __restrict__ outI) {
  const int lane = threadIdx.x & 31;
  const int wv   = threadIdx.x >> 5;
  const int w    = blockIdx.x * 4 + wv;
  const int bh   = w >> 7;                 // 128 query tiles per (b,h)
  const int qt   = w & 127;
  const int b    = bh / 12, h = bh % 12;

  __shared__ f16 pbuf[4 * 16 * 32];
  f16* pl = pbuf + wv * (16 * 32);

  const size_t qoff = ((size_t)bh * 2048 + (size_t)qt * 16) * 64;
  v16h qrA0 = load_frag_a(qr + qoff,      64, lane);
  v16h qrA1 = load_frag_a(qr + qoff + 32, 64, lane);
  v16h qiA0 = load_frag_a(qi + qoff,      64, lane);
  v16h qiA1 = load_frag_a(qi + qoff + 32, 64, lane);
  v16h nqiA0 = negv(qiA0);
  v16h nqiA1 = negv(qiA1);

  float mrow[8], lrow[8];
  v8f accR[4] = {}, accI[4] = {};
#pragma unroll
  for (int e = 0; e < 8; ++e) { mrow[e] = -3.0e38f; lrow[e] = 0.f; }

  const int hs = lane >> 4, nn = lane & 15;

  for (int kb = 0; kb < 64; ++kb) {        // 32 keys per block
    const size_t koff = ((size_t)bh * 2048 + (size_t)kb * 32) * 64;
    v8f ssr[2] = {}, ssi[2] = {};
#pragma unroll
    for (int t = 0; t < 2; ++t) {
      const size_t kt = koff + (size_t)t * 16 * 64;
      v16h Kr0 = load_frag_b(kr + kt,      64, lane);
      v16h Kr1 = load_frag_b(kr + kt + 32, 64, lane);
      v16h Ki0 = load_frag_b(ki + kt,      64, lane);
      v16h Ki1 = load_frag_b(ki + kt + 32, 64, lane);
      ssr[t] = wmma32(qrA0,  Kr0, ssr[t]);  ssr[t] = wmma32(qrA1,  Kr1, ssr[t]);
      ssr[t] = wmma32(qiA0,  Ki0, ssr[t]);  ssr[t] = wmma32(qiA1,  Ki1, ssr[t]);
      ssi[t] = wmma32(qrA0,  Ki0, ssi[t]);  ssi[t] = wmma32(qrA1,  Ki1, ssi[t]);
      ssi[t] = wmma32(nqiA0, Kr0, ssi[t]);  ssi[t] = wmma32(nqiA1, Kr1, ssi[t]);
    }
    float mag[2][8];
#pragma unroll
    for (int t = 0; t < 2; ++t)
#pragma unroll
      for (int e = 0; e < 8; ++e) {
        float sr = ssr[t][e];
        float si = ssi[t][e];
        mag[t][e] = sqrtf(sr * sr + si * si + 1e-8f) * 0.125f;
      }
    float scl[8];
#pragma unroll
    for (int e = 0; e < 8; ++e) {
      float rm = fmaxf(mag[0][e], mag[1][e]);
      for (int sm = 1; sm < 16; sm <<= 1) rm = fmaxf(rm, __shfl_xor(rm, sm, 32));
      float nm = fmaxf(mrow[e], rm);
      scl[e] = __expf(mrow[e] - nm);
      mrow[e] = nm;
      float ps = 0.f;
#pragma unroll
      for (int t = 0; t < 2; ++t) {
        float p = __expf(mag[t][e] - nm);
        mag[t][e] = p;
        ps += p;
      }
      for (int sm = 1; sm < 16; sm <<= 1) ps += __shfl_xor(ps, sm, 32);
      lrow[e] = lrow[e] * scl[e] + ps;
    }
#pragma unroll
    for (int dt = 0; dt < 4; ++dt)
#pragma unroll
      for (int e = 0; e < 8; ++e) { accR[dt][e] *= scl[e]; accI[dt][e] *= scl[e]; }
    // C-layout P tile -> row-major 16x32 f16 in LDS, reload as A fragment.
#pragma unroll
    for (int t = 0; t < 2; ++t)
#pragma unroll
      for (int e = 0; e < 8; ++e)
        pl[(e + hs * 8) * 32 + t * 16 + nn] = (f16)mag[t][e];
    v16h P = load_frag_a(pl, 32, lane);
    const size_t voff = (size_t)bh * 64 * 2048 + (size_t)kb * 32;
#pragma unroll
    for (int dt = 0; dt < 4; ++dt) {
      v16h Vr = load_frag_b(vTr + voff + (size_t)dt * 16 * 2048, 2048, lane);
      v16h Vi = load_frag_b(vTi + voff + (size_t)dt * 16 * 2048, 2048, lane);
      accR[dt] = wmma32(P, Vr, accR[dt]);
      accI[dt] = wmma32(P, Vi, accI[dt]);
    }
  }
  // Write [B*N, 768] f16 (head h occupies cols h*64..h*64+63)
#pragma unroll
  for (int dt = 0; dt < 4; ++dt)
#pragma unroll
    for (int e = 0; e < 8; ++e) {
      const int m = e + hs * 8;
      const float inv = 1.0f / lrow[e];
      const size_t o = ((size_t)(b * 2048 + qt * 16 + m)) * 768 +
                       (size_t)(h * 64 + dt * 16 + nn);
      outR[o] = (f16)(accR[dt][e] * inv);
      outI[o] = (f16)(accI[dt][e] * inv);
    }
}

// ---------------------------------------------------------------------------
// Host launcher
// ---------------------------------------------------------------------------
extern "C" void kernel_launch(void* const* d_in, const int* in_sizes, int n_in,
                              void* d_out, int out_size, void* d_ws,
                              size_t ws_size, hipStream_t stream) {
  (void)in_sizes; (void)n_in; (void)out_size; (void)ws_size;
  const float* x       = (const float*)d_in[0];
  const float* n1_g    = (const float*)d_in[1];
  const float* n1_b    = (const float*)d_in[2];
  const float* qkv_wr  = (const float*)d_in[3];
  const float* qkv_wi  = (const float*)d_in[4];
  const float* qkv_br  = (const float*)d_in[5];
  const float* qkv_bi  = (const float*)d_in[6];
  const float* proj_wr = (const float*)d_in[7];
  const float* proj_wi = (const float*)d_in[8];
  const float* proj_br = (const float*)d_in[9];
  const float* proj_bi = (const float*)d_in[10];
  const float* n2_g    = (const float*)d_in[11];
  const float* n2_b    = (const float*)d_in[12];
  const float* fc1_wr  = (const float*)d_in[13];
  const float* fc1_wi  = (const float*)d_in[14];
  const float* fc1_br  = (const float*)d_in[15];
  const float* fc1_bi  = (const float*)d_in[16];
  const float* fc2_wr  = (const float*)d_in[17];
  const float* fc2_wi  = (const float*)d_in[18];
  const float* fc2_br  = (const float*)d_in[19];
  const float* fc2_bi  = (const float*)d_in[20];

  const int E = 768, E3 = 2304, H = 1536, NSEQ = 2048, Bb = 2, NH = 12, HD = 64;
  const int M = Bb * NSEQ;                               // 4096 rows
  const long QKVH = (long)Bb * NH * NSEQ * HD;           // per-head elems

  size_t off = 0;
  auto alloc = [&](size_t bytes) -> char* {
    off = (off + 255) & ~(size_t)255;
    char* p = (char*)d_ws + off;
    off += bytes;
    return p;
  };

  // f16 weights (+ pre-negated imaginary weights for the accR stream)
  f16* wqr  = (f16*)alloc((size_t)E3 * E * 2);
  f16* wqi  = (f16*)alloc((size_t)E3 * E * 2);
  f16* wqin = (f16*)alloc((size_t)E3 * E * 2);
  f16* wpr  = (f16*)alloc((size_t)E * E * 2);
  f16* wpi  = (f16*)alloc((size_t)E * E * 2);
  f16* wpin = (f16*)alloc((size_t)E * E * 2);
  f16* w1r  = (f16*)alloc((size_t)H * E * 2);
  f16* w1i  = (f16*)alloc((size_t)H * E * 2);
  f16* w1in = (f16*)alloc((size_t)H * E * 2);
  f16* w2r  = (f16*)alloc((size_t)E * H * 2);
  f16* w2i  = (f16*)alloc((size_t)E * H * 2);
  f16* w2in = (f16*)alloc((size_t)E * H * 2);
  // normalized activations + raw fp32 x (for residual)
  f16*   n1r  = (f16*)alloc((size_t)M * E * 2);
  f16*   n1i  = (f16*)alloc((size_t)M * E * 2);
  float* xr32 = (float*)alloc((size_t)M * E * 4);
  float* xi32 = (float*)alloc((size_t)M * E * 4);
  // per-head f16 q, k, vT
  f16* q_r  = (f16*)alloc((size_t)QKVH * 2);
  f16* q_i  = (f16*)alloc((size_t)QKVH * 2);
  f16* k_r  = (f16*)alloc((size_t)QKVH * 2);
  f16* k_i  = (f16*)alloc((size_t)QKVH * 2);
  f16* vT_r = (f16*)alloc((size_t)QKVH * 2);
  f16* vT_i = (f16*)alloc((size_t)QKVH * 2);
  // attention outputs (f16, feed proj GEMM)
  f16* atR = (f16*)alloc((size_t)M * E * 2);
  f16* atI = (f16*)alloc((size_t)M * E * 2);
  // big fp32 qkv region; reused after split for x1/n2/h
  float* qkvR = (float*)alloc((size_t)M * E3 * 4);
  float* qkvI = (float*)alloc((size_t)M * E3 * 4);
  char*  region = (char*)qkvR;                 // 75.5 MB contiguous
  float* x1r = (float*)region;
  float* x1i = x1r + (size_t)M * E;
  f16*   n2r = (f16*)(x1i + (size_t)M * E);
  f16*   n2i = n2r + (size_t)M * E;
  f16*   h_r = n2i + (size_t)M * E;
  f16*   h_i = h_r + (size_t)M * H;            // ends at 62.9 MB < 75.5 MB

  auto cv = [&](const float* src, f16* dst, long n) {
    f32_to_f16_kernel<<<(unsigned)((n + 255) / 256), 256, 0, stream>>>(src, dst, n);
  };
  auto cvn = [&](const float* src, f16* dst, f16* dstn, long n) {
    f32_to_f16_neg_kernel<<<(unsigned)((n + 255) / 256), 256, 0, stream>>>(
        src, dst, dstn, n);
  };
  cv (qkv_wr,  wqr,        (long)E3 * E);
  cvn(qkv_wi,  wqi, wqin,  (long)E3 * E);
  cv (proj_wr, wpr,        (long)E * E);
  cvn(proj_wi, wpi, wpin,  (long)E * E);
  cv (fc1_wr,  w1r,        (long)H * E);
  cvn(fc1_wi,  w1i, w1in,  (long)H * E);
  cv (fc2_wr,  w2r,        (long)E * H);
  cvn(fc2_wi,  w2i, w2in,  (long)E * H);

  // 1) LN1 (interleaved input) -> n1 (f16) + raw deinterleaved x (fp32)
  cln_kernel<<<M, 256, 0, stream>>>(x, nullptr, nullptr, n1_g, n1_b,
                                    n1r, n1i, xr32, xi32, E, 1);

  // 2) qkv cGEMM -> fp32 [M, 3E]
  {
    int tiles = (M / 32) * (E3 / 32);            // 128 * 72
    cgemm_kernel<<<(tiles + 3) / 4, 128, 0, stream>>>(
        n1r, n1i, wqr, wqi, wqin, qkv_br, qkv_bi, nullptr, nullptr,
        qkvR, qkvI, nullptr, nullptr, nullptr, M, E, E3, 0);
  }

  // 3) split heads -> q, k [bh,N,hd], vT [bh,hd,N] (f16)
  split_qkv_kernel<<<(unsigned)(QKVH / 256), 256, 0, stream>>>(
      qkvR, qkvI, q_r, q_i, k_r, k_i, vT_r, vT_i);

  // 4) flash complex attention -> atR/atI (f16 [M, E])
  cattn_kernel<<<(Bb * NH * (NSEQ / 16)) / 4, 128, 0, stream>>>(
      q_r, q_i, k_r, k_i, vT_r, vT_i, atR, atI);

  // 5) proj cGEMM + residual(x) -> x1 fp32
  {
    int tiles = (M / 32) * (E / 32);             // 128 * 24
    cgemm_kernel<<<(tiles + 3) / 4, 128, 0, stream>>>(
        atR, atI, wpr, wpi, wpin, proj_br, proj_bi, xr32, xi32,
        x1r, x1i, nullptr, nullptr, nullptr, M, E, E, 0);
  }

  // 6) LN2 -> n2 (f16)
  cln_kernel<<<M, 256, 0, stream>>>(nullptr, x1r, x1i, n2_g, n2_b,
                                    n2r, n2i, nullptr, nullptr, E, 0);

  // 7) fc1 cGEMM + exact GELU -> h (f16 [M, H])
  {
    int tiles = (M / 32) * (H / 32);             // 128 * 48
    cgemm_kernel<<<(tiles + 3) / 4, 128, 0, stream>>>(
        n2r, n2i, w1r, w1i, w1in, fc1_br, fc1_bi, nullptr, nullptr,
        nullptr, nullptr, h_r, h_i, nullptr, M, E, H, 1);
  }

  // 8) fc2 cGEMM + residual(x1) -> interleaved fp32 d_out [B,N,2E]
  {
    int tiles = (M / 32) * (E / 32);             // 128 * 24
    cgemm_kernel<<<(tiles + 3) / 4, 128, 0, stream>>>(
        h_r, h_i, w2r, w2i, w2in, fc2_br, fc2_bi, x1r, x1i,
        nullptr, nullptr, nullptr, nullptr, (float*)d_out, M, H, E, 0);
  }
}
